// GAT_62036507623760
// MI455X (gfx1250) — matrix operand order repaired and checked by reading, compiled, not run on hardware
//
#include <hip/hip_runtime.h>
#include <hip/hip_bf16.h>
#include <math.h>

// ---------------------------------------------------------------------------
// Types for CDNA5 WMMA
// ---------------------------------------------------------------------------
typedef __bf16 bf16_t;
typedef __attribute__((ext_vector_type(16))) __bf16 v16bf;
typedef __attribute__((ext_vector_type(8)))  float  v8f;

#define GAT_N_NODES 10000
#define GAT_N_EDGES 100000

__device__ __forceinline__ unsigned short f32_to_bf16_rne(float f) {
  unsigned int u = __float_as_uint(f);
  u += 0x7fffu + ((u >> 16) & 1u);   // round-to-nearest-even
  return (unsigned short)(u >> 16);
}

// ---------------------------------------------------------------------------
// Utility kernels
// ---------------------------------------------------------------------------
__global__ void fill_f32(float* p, float v, size_t n) {
  size_t i = (size_t)blockIdx.x * blockDim.x + threadIdx.x;
  size_t stride = (size_t)gridDim.x * blockDim.x;
  for (; i < n; i += stride) p[i] = v;
}

// src: f32 [M,K] row-major -> dst: bf16 [M,Kpad] (padding pre-zeroed)
__global__ void cvt_pad_bf16(const float* __restrict__ src,
                             unsigned short* __restrict__ dst,
                             int M, int K, int Kpad) {
  size_t n = (size_t)M * K;
  size_t i = (size_t)blockIdx.x * blockDim.x + threadIdx.x;
  size_t stride = (size_t)gridDim.x * blockDim.x;
  for (; i < n; i += stride) {
    size_t m = i / (size_t)K;
    size_t k = i - m * (size_t)K;
    dst[m * (size_t)Kpad + k] = f32_to_bf16_rne(src[i]);
  }
}

// ---------------------------------------------------------------------------
// WMMA bf16 GEMM: C[M,N] f32 = A[M,Kpad] bf16 @ B[Kpad,N] bf16
// Block: 256 threads = 8 wave32 waves. Block tile 64(M) x 128(N) x 32(K).
// Wave tile: 16(M) x 64(N) -> 4 x v_wmma_f32_16x16x32_bf16 per K-step.
// B is staged TRANSPOSED in LDS so each lane's fragment is a contiguous,
// 16B-aligned 32-byte run -> 2x ds_load_b128 (instead of 16x ds_load_u16).
// ---------------------------------------------------------------------------
#define BM 64
#define BN 128
#define BK 32
#define BKP (BK + 8)   // padded row stride (80 bytes): banks spread, 16B aligned

__global__ __launch_bounds__(256)
void wmma_gemm_bf16(const unsigned short* __restrict__ A,
                    const unsigned short* __restrict__ B,
                    float* __restrict__ C,
                    int M, int N, int Kpad) {
  __shared__ __align__(16) unsigned short As[BM * BK];   // 64x32 bf16 = 4 KB
  __shared__ __align__(16) unsigned short Bst[BN][BKP];  // 128x(32+8) bf16 = 10 KB

  const int tid  = threadIdx.x;
  const int lane = tid & 31;
  const int wave = tid >> 5;     // 0..7
  const int wr   = wave & 3;     // 4 row groups of 16 rows
  const int wc   = wave >> 2;    // 2 col groups of 64 cols
  const int bm   = blockIdx.y * BM;
  const int bn   = blockIdx.x * BN;

  v8f acc[4] = {};

  for (int k0 = 0; k0 < Kpad; k0 += BK) {
    // ---- stage A tile (2048 elems, 8 bf16 per thread, one 16B vector) ----
    {
      int e = tid * 8;
      int r = e / BK;
      int c = e - r * BK;
      int gm = bm + r;
      unsigned short* dstp = &As[e];
      if (gm < M) {
        const unsigned short* src = A + (size_t)gm * Kpad + (k0 + c);
        *(uint4*)dstp = *(const uint4*)src;
        if (k0 + BK < Kpad) __builtin_prefetch(src + BK, 0, 0); // global_prefetch_b8
      } else {
        uint4 z = {0u, 0u, 0u, 0u};
        *(uint4*)dstp = z;
      }
    }
    // ---- stage B tile transposed: thread = 2(k) x 8(n) sub-block ----
    {
      int rp = tid >> 4;          // row-pair index 0..15  (k = 2*rp, 2*rp+1)
      int cg = tid & 15;          // col group 0..15
      int n0 = cg * 8;
      int gn0 = bn + n0;
      int k = k0 + rp * 2;
      const unsigned short* s0 = B + (size_t)k * N + gn0;
      const unsigned short* s1 = s0 + N;
      unsigned short lo[8], hi[8];
      if (gn0 + 7 < N) {
        *(uint4*)lo = *(const uint4*)s0;
        *(uint4*)hi = *(const uint4*)s1;
        if (k0 + BK < Kpad) __builtin_prefetch(s0 + (size_t)BK * N, 0, 0);
      } else {
        for (int i = 0; i < 8; ++i) {
          lo[i] = (gn0 + i < N) ? s0[i] : (unsigned short)0;
          hi[i] = (gn0 + i < N) ? s1[i] : (unsigned short)0;
        }
      }
      #pragma unroll
      for (int i = 0; i < 8; ++i) {
        unsigned int v = (unsigned int)lo[i] | ((unsigned int)hi[i] << 16);
        *(unsigned int*)&Bst[n0 + i][rp * 2] = v;   // one b32 per (n, k-pair)
      }
    }
    __syncthreads();

    const bf16_t* Asb = (const bf16_t*)As;
    const int l  = lane & 15;
    const int hi = lane >> 4;

    // A fragment (16x32 bf16), contiguous K order matching B:
    //   lane half 0: K = 0..15 ; lane half 1: K = 16..31
    v16bf af;
    {
      const bf16_t* arow = Asb + (wr * 16 + l) * BK;
      #pragma unroll
      for (int i = 0; i < 16; ++i) af[i] = arow[hi * 16 + i];
    }

    #pragma unroll
    for (int nb = 0; nb < 4; ++nb) {
      // B fragment (32x16 bf16): column N = lane&15, K = hi*16 .. hi*16+15
      // -> contiguous 32B run in transposed LDS tile (2x ds_load_b128)
      const int ncol = wc * 64 + nb * 16 + l;
      const bf16_t* bp = (const bf16_t*)&Bst[ncol][hi * 16];
      v16bf bf;
      #pragma unroll
      for (int k = 0; k < 16; ++k) bf[k] = bp[k];
      acc[nb] = __builtin_amdgcn_wmma_f32_16x16x32_bf16(
          /*neg_a=*/false, af, /*neg_b=*/false, bf,
          /*c_mod=*/(short)0, acc[nb], /*reuse_a=*/false, /*reuse_b=*/false);
    }
    __syncthreads();
  }

  // ---- store: C/D layout -> VGPR r holds M = r (lanes 0-15) / M = 8+r (16-31)
  const int l    = lane & 15;
  const int moff = (lane >> 4) ? 8 : 0;
  #pragma unroll
  for (int nb = 0; nb < 4; ++nb) {
    int gn = bn + wc * 64 + nb * 16 + l;
    if (gn >= N) continue;
    #pragma unroll
    for (int r = 0; r < 8; ++r) {
      int gm = bm + wr * 16 + moff + r;
      if (gm < M) C[(size_t)gm * N + gn] = acc[nb][r];
    }
  }
}

// ---------------------------------------------------------------------------
// Attention kernels
// ---------------------------------------------------------------------------
// One block per (node, head): s[n,h] = h[n,h,:].a_src[h,:], d likewise.
__global__ __launch_bounds__(256)
void attn_scores(const float* __restrict__ h,
                 const float* __restrict__ a_src, const float* __restrict__ a_dst,
                 float* __restrict__ s, float* __restrict__ d,
                 int Nn, int H, int C) {
  int nh = blockIdx.x;
  int node = nh / H, head = nh - node * H;
  const float* row = h + (size_t)node * H * C + (size_t)head * C;
  const float* as = a_src + head * C;
  const float* ad = a_dst + head * C;
  float ss = 0.f, dd = 0.f;
  for (int c = threadIdx.x; c < C; c += blockDim.x) {
    float v = row[c];
    ss += v * as[c];
    dd += v * ad[c];
  }
  __shared__ float red0[256];
  __shared__ float red1[256];
  red0[threadIdx.x] = ss;
  red1[threadIdx.x] = dd;
  __syncthreads();
  for (int st = 128; st > 0; st >>= 1) {
    if (threadIdx.x < st) {
      red0[threadIdx.x] += red0[threadIdx.x + st];
      red1[threadIdx.x] += red1[threadIdx.x + st];
    }
    __syncthreads();
  }
  if (threadIdx.x == 0) { s[nh] = red0[0]; d[nh] = red1[0]; }
}

__device__ __forceinline__ void atomicMaxF32(float* addr, float val) {
  unsigned int* ua = (unsigned int*)addr;
  unsigned int old = *ua;
  while (__uint_as_float(old) < val) {
    unsigned int assumed = old;
    old = atomicCAS(ua, assumed, __float_as_uint(val));
    if (old == assumed) break;
  }
}

// e[E*H] = leaky_relu(s[src]+d[dst]); m[dst,h] = segment max
__global__ void edge_logits_max(const int* __restrict__ src, const int* __restrict__ dst,
                                const float* __restrict__ s, const float* __restrict__ d,
                                float* __restrict__ e, float* __restrict__ m,
                                int E, int H) {
  int i = blockIdx.x * blockDim.x + threadIdx.x;
  if (i >= E * H) return;
  int ei = i / H, h = i - ei * H;
  int u = src[ei], v = dst[ei];
  float x = s[u * H + h] + d[v * H + h];
  x = x > 0.f ? x : 0.2f * x;
  e[i] = x;
  atomicMaxF32(&m[v * H + h], x);
}

// reference: m = where(isfinite(m), m, 0)
__global__ void fix_nonfinite(float* m, int n) {
  int i = blockIdx.x * blockDim.x + threadIdx.x;
  if (i >= n) return;
  float v = m[i];
  if (!(fabsf(v) <= 3.0e38f)) m[i] = 0.f;
}

// p = exp(e - m[dst]) (in place over e); z[dst,h] += p
__global__ void edge_exp_sum(const int* __restrict__ dst,
                             float* __restrict__ e, const float* __restrict__ m,
                             float* __restrict__ z, int E, int H) {
  int i = blockIdx.x * blockDim.x + threadIdx.x;
  if (i >= E * H) return;
  int ei = i / H, h = i - ei * H;
  int v = dst[ei];
  float p = __expf(e[i] - m[v * H + h]);
  e[i] = p;
  atomicAdd(&z[v * H + h], p);
}

// out[dst, h, :] += (p / (z[dst,h]+1e-16)) * hfeat[src, h, :]
__global__ __launch_bounds__(256)
void scatter_agg(const int* __restrict__ src, const int* __restrict__ dst,
                 const float* __restrict__ p, const float* __restrict__ z,
                 const float* __restrict__ hfeat, float* __restrict__ out,
                 int E, int H, int C) {
  int ei = blockIdx.x;
  int u = src[ei], v = dst[ei];
  int HC = H * C;
  for (int j = threadIdx.x; j < HC; j += blockDim.x) {
    int head = j / C;
    float alpha = p[ei * H + head] / (z[v * H + head] + 1e-16f);
    atomicAdd(&out[(size_t)v * HC + j], alpha * hfeat[(size_t)u * HC + j]);
  }
}

// layer-1 epilogue: agg + bias, ELU, convert to padded bf16 [Nrows, Kpad]
__global__ void bias_elu_to_bf16(const float* __restrict__ agg,
                                 const float* __restrict__ bias,
                                 unsigned short* __restrict__ dst,
                                 int Nrows, int HC, int Kpad) {
  size_t n = (size_t)Nrows * HC;
  size_t i = (size_t)blockIdx.x * blockDim.x + threadIdx.x;
  size_t stride = (size_t)gridDim.x * blockDim.x;
  for (; i < n; i += stride) {
    size_t row = i / (size_t)HC;
    size_t col = i - row * (size_t)HC;
    float v = agg[i] + bias[col];
    v = v > 0.f ? v : (__expf(v) - 1.f);   // ELU(alpha=1)
    dst[row * (size_t)Kpad + col] = f32_to_bf16_rne(v);
  }
}

// layer-2 epilogue: out = agg + bias (H=1, mean over heads == identity)
__global__ void add_bias_out(const float* __restrict__ agg,
                             const float* __restrict__ bias,
                             float* __restrict__ out, int Nrows, int C) {
  int n = Nrows * C;
  int i = blockIdx.x * blockDim.x + threadIdx.x;
  if (i >= n) return;
  out[i] = agg[i] + bias[i % C];
}

// ---------------------------------------------------------------------------
// Host-side orchestration
// ---------------------------------------------------------------------------
extern "C" void kernel_launch(void* const* d_in, const int* in_sizes, int n_in,
                              void* d_out, int out_size, void* d_ws, size_t ws_size,
                              hipStream_t stream) {
  const float* x        = (const float*)d_in[0];   // [10000,500]
  const int*   eidx     = (const int*)  d_in[1];   // [2,100000]
  const float* W1       = (const float*)d_in[2];   // [500,8000]
  const float* att_src1 = (const float*)d_in[3];   // [8,1000]
  const float* att_dst1 = (const float*)d_in[4];   // [8,1000]
  const float* b1       = (const float*)d_in[5];   // [8000]
  const float* W2       = (const float*)d_in[6];   // [8000,200]
  const float* att_src2 = (const float*)d_in[7];   // [1,200]
  const float* att_dst2 = (const float*)d_in[8];   // [1,200]
  const float* b2       = (const float*)d_in[9];   // [200]
  float* out = (float*)d_out;                      // [10000,200]

  const int N = GAT_N_NODES, E = GAT_N_EDGES;
  const int K1 = 500, K1p = 512, HC1 = 8000, H1 = 8, C1 = 1000;
  const int K2 = 8000, K2p = 8192, C2 = 200;
  const int* src = eidx;
  const int* dst = eidx + E;

  // Workspace carve-up (256B aligned)
  size_t off = 0;
  auto carve = [&](size_t bytes) -> char* {
    char* p = (char*)d_ws + off;
    off += (bytes + 255) & ~(size_t)255;
    return p;
  };
  unsigned short* xb   = (unsigned short*)carve((size_t)N   * K1p * 2); // x bf16 padded
  unsigned short* w1b  = (unsigned short*)carve((size_t)K1p * HC1 * 2); // W1 bf16 padded
  float* h1   = (float*)carve((size_t)N * HC1 * 4);   // x @ W1
  float* agg1 = (float*)carve((size_t)N * HC1 * 4);   // layer1 aggregation
  float* s1   = (float*)carve((size_t)N * H1 * 4);
  float* d1   = (float*)carve((size_t)N * H1 * 4);
  float* m1   = (float*)carve((size_t)N * H1 * 4);
  float* z1   = (float*)carve((size_t)N * H1 * 4);
  float* el1  = (float*)carve((size_t)E * H1 * 4);    // logits then p
  unsigned short* heb = (unsigned short*)carve((size_t)N * K2p * 2); // elu(h) bf16 padded
  unsigned short* w2b = (unsigned short*)carve((size_t)K2p * C2 * 2);
  float* h2   = (float*)carve((size_t)N * C2 * 4);
  float* agg2 = (float*)carve((size_t)N * C2 * 4);
  float* s2   = (float*)carve((size_t)N * 4);
  float* d2   = (float*)carve((size_t)N * 4);
  float* m2   = (float*)carve((size_t)N * 4);
  float* z2   = (float*)carve((size_t)N * 4);
  float* el2  = (float*)carve((size_t)E * 4);

  const int TB = 256;
  const int FG = 4096; // grid-stride fill/convert grid

  // ---- zero / init (bf16 buffers zeroed as 0x0000 via f32 fill) ----
  fill_f32<<<FG, TB, 0, stream>>>((float*)xb,  0.f, (size_t)N * K1p / 2);
  fill_f32<<<FG, TB, 0, stream>>>((float*)w1b, 0.f, (size_t)K1p * HC1 / 2);
  fill_f32<<<FG, TB, 0, stream>>>((float*)heb, 0.f, (size_t)N * K2p / 2);
  fill_f32<<<FG, TB, 0, stream>>>((float*)w2b, 0.f, (size_t)K2p * C2 / 2);
  fill_f32<<<FG, TB, 0, stream>>>(agg1, 0.f, (size_t)N * HC1);
  fill_f32<<<FG, TB, 0, stream>>>(agg2, 0.f, (size_t)N * C2);
  fill_f32<<<FG, TB, 0, stream>>>(z1, 0.f, (size_t)N * H1);
  fill_f32<<<FG, TB, 0, stream>>>(z2, 0.f, (size_t)N);
  fill_f32<<<FG, TB, 0, stream>>>(m1, -INFINITY, (size_t)N * H1);
  fill_f32<<<FG, TB, 0, stream>>>(m2, -INFINITY, (size_t)N);

  // ---- layer 1 ----
  cvt_pad_bf16<<<FG, TB, 0, stream>>>(x,  xb,  N,  K1, K1p);
  cvt_pad_bf16<<<FG, TB, 0, stream>>>(W1, w1b, K1, HC1, HC1);

  dim3 g1((HC1 + BN - 1) / BN, (N + BM - 1) / BM);
  wmma_gemm_bf16<<<g1, TB, 0, stream>>>(xb, w1b, h1, N, HC1, K1p);

  attn_scores<<<N * H1, TB, 0, stream>>>(h1, att_src1, att_dst1, s1, d1, N, H1, C1);
  edge_logits_max<<<(E * H1 + TB - 1) / TB, TB, 0, stream>>>(src, dst, s1, d1, el1, m1, E, H1);
  fix_nonfinite<<<(N * H1 + TB - 1) / TB, TB, 0, stream>>>(m1, N * H1);
  edge_exp_sum<<<(E * H1 + TB - 1) / TB, TB, 0, stream>>>(dst, el1, m1, z1, E, H1);
  scatter_agg<<<E, TB, 0, stream>>>(src, dst, el1, z1, h1, agg1, E, H1, C1);
  bias_elu_to_bf16<<<FG, TB, 0, stream>>>(agg1, b1, heb, N, HC1, K2p);

  // ---- layer 2 ----
  cvt_pad_bf16<<<FG, TB, 0, stream>>>(W2, w2b, K2, C2, C2);
  dim3 g2((C2 + BN - 1) / BN, (N + BM - 1) / BM);
  wmma_gemm_bf16<<<g2, TB, 0, stream>>>(heb, w2b, h2, N, C2, K2p);

  attn_scores<<<N, TB, 0, stream>>>(h2, att_src2, att_dst2, s2, d2, N, 1, C2);
  edge_logits_max<<<(E + TB - 1) / TB, TB, 0, stream>>>(src, dst, s2, d2, el2, m2, E, 1);
  fix_nonfinite<<<(N + TB - 1) / TB, TB, 0, stream>>>(m2, N);
  edge_exp_sum<<<(E + TB - 1) / TB, TB, 0, stream>>>(dst, el2, m2, z2, E, 1);
  scatter_agg<<<E, TB, 0, stream>>>(src, dst, el2, z2, h2, agg2, E, 1, C2);
  add_bias_out<<<(N * C2 + TB - 1) / TB, TB, 0, stream>>>(agg2, b2, out, N, C2);
}
